// GlobalDistNet_40157944217635
// MI455X (gfx1250) — matrix operand order repaired and checked by reading, not compiled, not run on hardware
//
#include <hip/hip_runtime.h>
#include <hip/hip_bf16.h>

// ---------------------------------------------------------------------------
// Types for CDNA5 WMMA (wave32, 16x16x32 bf16 -> f32)
// ---------------------------------------------------------------------------
typedef __bf16 bf16_t;
typedef __attribute__((ext_vector_type(8)))  float  v8f;
typedef __attribute__((ext_vector_type(8)))  bf16_t v8bf;
typedef __attribute__((ext_vector_type(16))) bf16_t v16bf;

// Optional CDNA5 async global->LDS path (tracked by ASYNCcnt)
#if defined(__has_builtin)
#if __has_builtin(__builtin_amdgcn_global_load_async_to_lds_b128) && \
    __has_builtin(__builtin_amdgcn_s_wait_asynccnt)
#define USE_ASYNC_LDS 1
#endif
#endif
#ifndef USE_ASYNC_LDS
#define USE_ASYNC_LDS 0
#endif

#if USE_ASYNC_LDS
#define AS1 __attribute__((address_space(1)))
#define AS3 __attribute__((address_space(3)))
typedef int i32x4 __attribute__((__vector_size__(16)));
__device__ __forceinline__ AS1 i32x4* to_global(const void* p) {
  return (AS1 i32x4*)(unsigned long long)(uintptr_t)p;
}
// flat LDS addresses carry the wave-relative LDS byte offset in the low 32 bits
__device__ __forceinline__ AS3 i32x4* to_lds(const void* p) {
  return (AS3 i32x4*)(unsigned)(uintptr_t)p;
}
#endif

__device__ __forceinline__ unsigned short f2bf(float f) {
  unsigned u = __float_as_uint(f);
  u += 0x7fffu + ((u >> 16) & 1u);   // round-to-nearest-even
  return (unsigned short)(u >> 16);
}
__device__ __forceinline__ float lrelu(float v, float s) { return v >= 0.f ? v : s * v; }
// order-preserving float<->uint mapping for atomicMax on floats
__device__ __forceinline__ unsigned fenc(float f) {
  unsigned u = __float_as_uint(f);
  return (u >> 31) ? ~u : (u ^ 0x80000000u);
}
__device__ __forceinline__ float fdec(unsigned u) {
  return __uint_as_float((u >> 31) ? (u ^ 0x80000000u) : ~u);
}

// ---------------------------------------------------------------------------
// Small utility kernels
// ---------------------------------------------------------------------------
__global__ void fill_f32_k(float* p, float v, int n) {
  int i = blockIdx.x * blockDim.x + threadIdx.x;
  if (i < n) p[i] = v;
}
__global__ void rsqrt_ip_k(float* p, int n) {
  int i = blockIdx.x * blockDim.x + threadIdx.x;
  if (i < n) p[i] = rsqrtf(p[i]);
}
__global__ void leaky_ip_k(float* p, long n, float s) {
  long i = (long)blockIdx.x * blockDim.x + threadIdx.x;
  if (i < n) p[i] = lrelu(p[i], s);
}
__global__ void deg_edges_k(const int* __restrict__ ei, float* deg, int E) {
  int e = blockIdx.x * blockDim.x + threadIdx.x;
  if (e < E) atomicAdd(&deg[ei[E + e]], 1.0f);
}
// W[K][128] (row-major f32) -> Bt[128][K] (bf16, "column-of-W per row")
__global__ void wtrans_k(const float* __restrict__ W, unsigned short* __restrict__ Bt, int K) {
  int idx = blockIdx.x * blockDim.x + threadIdx.x;
  if (idx < K * 128) {
    int k = idx >> 7, c = idx & 127;
    Bt[(long)c * K + k] = f2bf(W[idx]);
  }
}

// ---------------------------------------------------------------------------
// WMMA GEMM kernels. 256 threads = 8 waves; 128-row x 128-col block tile;
// K consumed in 32-wide chunks; double-buffered LDS; one barrier/iteration.
//   A LDS layout: [row][k] stride SA (16B-aligned fragment chunks)
//   B LDS layout: [col][k] stride SB (32B-aligned fragment loads)
// ---------------------------------------------------------------------------
#define SA 40   // 80B rows
#define SB 48   // 96B rows

// GEMM 1: h = assemble(emb[poi] || dist) @ W_in   (M=N, K=4128, Ncols=128)
__global__ __launch_bounds__(256)
void gemm_embed_k(const int* __restrict__ poi, const float* __restrict__ dist,
                  const float* __restrict__ emb, const unsigned short* __restrict__ Bt,
                  float* __restrict__ outb, int nrows) {
  __shared__ __attribute__((aligned(64))) unsigned short As[2][128 * SA];
  __shared__ __attribute__((aligned(64))) unsigned short Bs[2][128 * SB];
  __shared__ int Pois[128 * 32];

  const int tid = threadIdx.x;
  const int wave = tid >> 5, lane = tid & 31;
  const int hi = lane >> 4, ln = lane & 15;
  const int blockRow = blockIdx.x * 128;
  const int rowBase = wave * 16;
  const int srow = tid >> 1;        // staging row / column 0..127
  const int skb = (tid & 1) * 16;   // staging K sub-offset 0|16
  const int node = blockRow + srow;

  // stage this block's poi indices once (removes dependent gather from hot loop)
  for (int i = tid; i < 128 * 32; i += 256) {
    int row = i >> 5;
    int nd = blockRow + row;
    Pois[i] = (nd < nrows) ? poi[(long)nd * 32 + (i & 31)] : 0;
  }
  __syncthreads();

  v8f acc[8] = {};
  float av[16];

  auto loadA = [&](int kc) {
    if (node < nrows) {
      if (kc < 128) {
        int idx = Pois[srow * 32 + (kc >> 2)];
        const float* src = emb + (long)idx * 128 + (kc & 3) * 32 + skb;
#pragma unroll
        for (int i = 0; i < 16; i++) av[i] = src[i];
      } else {
        const float* src = dist + (long)node * 32 + skb;
#pragma unroll
        for (int i = 0; i < 16; i++) av[i] = src[i];
      }
    } else {
#pragma unroll
      for (int i = 0; i < 16; i++) av[i] = 0.f;
    }
  };
  auto storeA = [&](int buf) {
#pragma unroll
    for (int i = 0; i < 16; i++) As[buf][srow * SA + skb + i] = f2bf(av[i]);
  };
  auto stageB = [&](int kc, int buf) {
    const unsigned short* bsrc = Bt + (long)srow * 4128 + kc * 32 + skb;
    unsigned short* bdst = &Bs[buf][srow * SB + skb];
#if USE_ASYNC_LDS
    __builtin_amdgcn_global_load_async_to_lds_b128(to_global(bsrc), to_lds(bdst), 0, 0);
    __builtin_amdgcn_global_load_async_to_lds_b128(to_global(bsrc + 8), to_lds(bdst + 8), 0, 0);
#else
#pragma unroll
    for (int i = 0; i < 16; i++) bdst[i] = bsrc[i];
#endif
  };
  auto compute = [&](int buf) {
    const int c0 = hi * 16;
    const unsigned short* ap = &As[buf][(rowBase + ln) * SA + c0];
    v8bf alo = *reinterpret_cast<const v8bf*>(ap);
    v8bf ahi = *reinterpret_cast<const v8bf*>(ap + 8);
    v16bf afrag = __builtin_shufflevector(alo, ahi, 0, 1, 2, 3, 4, 5, 6, 7,
                                          8, 9, 10, 11, 12, 13, 14, 15);
#pragma unroll
    for (int ct = 0; ct < 8; ++ct) {
      const unsigned short* bp = &Bs[buf][(ct * 16 + ln) * SB + hi * 16];
      v16bf bfrag = *reinterpret_cast<const v16bf*>(bp);
      acc[ct] = __builtin_amdgcn_wmma_f32_16x16x32_bf16(
          false, afrag, false, bfrag, (short)0, acc[ct], false, false);
    }
  };

  // prologue: fill buffer 0
  loadA(0);
  stageB(0, 0);
  storeA(0);
#if USE_ASYNC_LDS
  __builtin_amdgcn_s_wait_asynccnt(0);
#endif
  __syncthreads();

  for (int kc = 0; kc < 129; ++kc) {   // 129 chunks of K=32 (4096 emb + 32 dist)
    const int cur = kc & 1, nxt = cur ^ 1;
    const bool more = (kc + 1) < 129;
    if (more) { loadA(kc + 1); stageB(kc + 1, nxt); }  // overlap with WMMA below
    compute(cur);
    if (more) {
      storeA(nxt);
#if USE_ASYNC_LDS
      __builtin_amdgcn_s_wait_asynccnt(0);
#endif
      __syncthreads();
    }
  }

  // C/D layout: VGPR r -> row r (+8 for lanes 16..31), col = lane&15
  const int grow0 = blockRow + rowBase + hi * 8;
#pragma unroll
  for (int ct = 0; ct < 8; ++ct)
#pragma unroll
    for (int r = 0; r < 8; r++) {
      int grow = grow0 + r;
      if (grow < nrows) outb[(long)grow * 128 + ct * 16 + ln] = acc[ct][r];
    }
}

// GEMM 2: h = x @ W   (M=N, K=128, Ncols=128), x f32 global, W bf16 transposed
__global__ __launch_bounds__(256)
void gemm_x_k(const float* __restrict__ x, const unsigned short* __restrict__ Bt,
              float* __restrict__ outb, int nrows) {
  __shared__ __attribute__((aligned(64))) unsigned short As[2][128 * SA];
  __shared__ __attribute__((aligned(64))) unsigned short Bs[2][128 * SB];

  const int tid = threadIdx.x;
  const int wave = tid >> 5, lane = tid & 31;
  const int hi = lane >> 4, ln = lane & 15;
  const int blockRow = blockIdx.x * 128;
  const int rowBase = wave * 16;
  const int srow = tid >> 1;
  const int skb = (tid & 1) * 16;
  const int node = blockRow + srow;

  v8f acc[8] = {};
  float av[16];

  auto loadA = [&](int kc) {
    if (node < nrows) {
      const float* src = x + (long)node * 128 + kc * 32 + skb;
#pragma unroll
      for (int i = 0; i < 16; i++) av[i] = src[i];
    } else {
#pragma unroll
      for (int i = 0; i < 16; i++) av[i] = 0.f;
    }
  };
  auto storeA = [&](int buf) {
#pragma unroll
    for (int i = 0; i < 16; i++) As[buf][srow * SA + skb + i] = f2bf(av[i]);
  };
  auto stageB = [&](int kc, int buf) {
    const unsigned short* bsrc = Bt + (long)srow * 128 + kc * 32 + skb;
    unsigned short* bdst = &Bs[buf][srow * SB + skb];
#if USE_ASYNC_LDS
    __builtin_amdgcn_global_load_async_to_lds_b128(to_global(bsrc), to_lds(bdst), 0, 0);
    __builtin_amdgcn_global_load_async_to_lds_b128(to_global(bsrc + 8), to_lds(bdst + 8), 0, 0);
#else
#pragma unroll
    for (int i = 0; i < 16; i++) bdst[i] = bsrc[i];
#endif
  };
  auto compute = [&](int buf) {
    const int c0 = hi * 16;
    const unsigned short* ap = &As[buf][(rowBase + ln) * SA + c0];
    v8bf alo = *reinterpret_cast<const v8bf*>(ap);
    v8bf ahi = *reinterpret_cast<const v8bf*>(ap + 8);
    v16bf afrag = __builtin_shufflevector(alo, ahi, 0, 1, 2, 3, 4, 5, 6, 7,
                                          8, 9, 10, 11, 12, 13, 14, 15);
#pragma unroll
    for (int ct = 0; ct < 8; ++ct) {
      const unsigned short* bp = &Bs[buf][(ct * 16 + ln) * SB + hi * 16];
      v16bf bfrag = *reinterpret_cast<const v16bf*>(bp);
      acc[ct] = __builtin_amdgcn_wmma_f32_16x16x32_bf16(
          false, afrag, false, bfrag, (short)0, acc[ct], false, false);
    }
  };

  loadA(0);
  stageB(0, 0);
  storeA(0);
#if USE_ASYNC_LDS
  __builtin_amdgcn_s_wait_asynccnt(0);
#endif
  __syncthreads();

#pragma unroll
  for (int kc = 0; kc < 4; ++kc) {
    const int cur = kc & 1, nxt = cur ^ 1;
    const bool more = (kc + 1) < 4;
    if (more) { loadA(kc + 1); stageB(kc + 1, nxt); }
    compute(cur);
    if (more) {
      storeA(nxt);
#if USE_ASYNC_LDS
      __builtin_amdgcn_s_wait_asynccnt(0);
#endif
      __syncthreads();
    }
  }

  const int grow0 = blockRow + rowBase + hi * 8;
#pragma unroll
  for (int ct = 0; ct < 8; ++ct)
#pragma unroll
    for (int r = 0; r < 8; r++) {
      int grow = grow0 + r;
      if (grow < nrows) outb[(long)grow * 128 + ct * 16 + ln] = acc[ct][r];
    }
}

// ---------------------------------------------------------------------------
// GCN scatter (deg^{-1/2} symmetric norm) : init with bias + self-loop term,
// then edge-parallel atomics (32 lanes/edge, 4 feats/lane).
// ---------------------------------------------------------------------------
__global__ void gcn_init_k(const float* __restrict__ h, const float* __restrict__ dinv,
                           const float* __restrict__ bias, float* __restrict__ outb, int n) {
  long idx = (long)blockIdx.x * blockDim.x + threadIdx.x;
  if (idx >= (long)n * 128) return;
  int i = (int)(idx >> 7), c = (int)(idx & 127);
  float di = dinv[i];
  outb[idx] = bias[c] + h[idx] * di * di;
}
__global__ void gcn_edges_k(const float* __restrict__ h, float* __restrict__ outb,
                            const int* __restrict__ ei, const float* __restrict__ dinv, int E) {
  long tid = (long)blockIdx.x * blockDim.x + threadIdx.x;
  int e = (int)(tid >> 5), l = (int)(tid & 31);
  if (e >= E) return;
  int s = ei[e], d = ei[E + e];
  float w = dinv[s] * dinv[d];
  const float4 v = *(const float4*)(h + (long)s * 128 + l * 4);
  float* o = outb + (long)d * 128 + l * 4;
  atomicAdd(o + 0, v.x * w); atomicAdd(o + 1, v.y * w);
  atomicAdd(o + 2, v.z * w); atomicAdd(o + 3, v.w * w);
}

// ---------------------------------------------------------------------------
// GAT: per-node dots, segment-max (uint-encoded), exp-sum, weighted scatter
// ---------------------------------------------------------------------------
__global__ void gat_dots_k(const float* __restrict__ h, const float* __restrict__ asv,
                           const float* __restrict__ adv, float* als, float* ald, int n) {
  long g = (long)blockIdx.x * blockDim.x + threadIdx.x;
  int node = (int)(g >> 5), l = (int)(g & 31);
  if (node >= n) return;
  const float* row = h + (long)node * 128;
  float a = 0.f, b = 0.f;
  for (int c = l; c < 128; c += 32) { float v = row[c]; a += v * asv[c]; b += v * adv[c]; }
  for (int off = 16; off; off >>= 1) { a += __shfl_down(a, off, 32); b += __shfl_down(b, off, 32); }
  if (l == 0) { als[node] = a; ald[node] = b; }
}
__global__ void emax_init_k(const float* als, const float* ald, unsigned* emaxu, int n) {
  int i = blockIdx.x * blockDim.x + threadIdx.x;
  if (i < n) emaxu[i] = fenc(lrelu(als[i] + ald[i], 0.2f));
}
__global__ void emax_edges_k(const float* als, const float* ald, unsigned* emaxu,
                             const int* __restrict__ ei, int E) {
  int e = blockIdx.x * blockDim.x + threadIdx.x;
  if (e >= E) return;
  int s = ei[e], d = ei[E + e];
  atomicMax(&emaxu[d], fenc(lrelu(als[s] + ald[d], 0.2f)));
}
__global__ void esum_init_k(const float* als, const float* ald, const unsigned* emaxu,
                            float* esum, int n) {
  int i = blockIdx.x * blockDim.x + threadIdx.x;
  if (i >= n) return;
  esum[i] = __expf(lrelu(als[i] + ald[i], 0.2f) - fdec(emaxu[i]));
}
__global__ void esum_edges_k(const float* als, const float* ald, const unsigned* emaxu,
                             float* esum, const int* __restrict__ ei, int E) {
  int e = blockIdx.x * blockDim.x + threadIdx.x;
  if (e >= E) return;
  int s = ei[e], d = ei[E + e];
  atomicAdd(&esum[d], __expf(lrelu(als[s] + ald[d], 0.2f) - fdec(emaxu[d])));
}
__global__ void gat_init_k(const float* __restrict__ h, const float* als, const float* ald,
                           const unsigned* emaxu, const float* esum,
                           const float* __restrict__ bias, float* __restrict__ outb, int n) {
  long idx = (long)blockIdx.x * blockDim.x + threadIdx.x;
  if (idx >= (long)n * 128) return;
  int i = (int)(idx >> 7), c = (int)(idx & 127);
  float a = __expf(lrelu(als[i] + ald[i], 0.2f) - fdec(emaxu[i])) / esum[i];
  outb[idx] = bias[c] + h[idx] * a;
}
__global__ void gat_edges_k(const float* __restrict__ h, const float* als, const float* ald,
                            const unsigned* emaxu, const float* esum,
                            const int* __restrict__ ei, float* __restrict__ outb, int E) {
  long tid = (long)blockIdx.x * blockDim.x + threadIdx.x;
  int e = (int)(tid >> 5), l = (int)(tid & 31);
  if (e >= E) return;
  int s = ei[e], d = ei[E + e];
  float a = __expf(lrelu(als[s] + ald[d], 0.2f) - fdec(emaxu[d])) / esum[d];
  const float4 v = *(const float4*)(h + (long)s * 128 + l * 4);
  float* o = outb + (long)d * 128 + l * 4;
  atomicAdd(o + 0, v.x * a); atomicAdd(o + 1, v.y * a);
  atomicAdd(o + 2, v.z * a); atomicAdd(o + 3, v.w * a);
}

// ---------------------------------------------------------------------------
// GraphNorm (node-axis mean/var per channel) + leaky + residual
// ---------------------------------------------------------------------------
__global__ void col_acc_k(const float* __restrict__ t, float* colsum, int n) {
  int c = threadIdx.x & 127, rl = threadIdx.x >> 7;
  float acc = 0.f;
  for (int i = blockIdx.x * 2 + rl; i < n; i += gridDim.x * 2) acc += t[(long)i * 128 + c];
  atomicAdd(&colsum[c], acc);
}
__global__ void colsq_acc_k(const float* __restrict__ t, const float* colsum,
                            const float* __restrict__ ms, float* colsq, int n) {
  int c = threadIdx.x & 127, rl = threadIdx.x >> 7;
  float sub = ms[c] * (colsum[c] / (float)n);
  float acc = 0.f;
  for (int i = blockIdx.x * 2 + rl; i < n; i += gridDim.x * 2) {
    float o = t[(long)i * 128 + c] - sub;
    acc += o * o;
  }
  atomicAdd(&colsq[c], acc);
}
__global__ void norm_res_k(float* __restrict__ x, const float* __restrict__ t,
                           const float* colsum, const float* colsq,
                           const float* __restrict__ w, const float* __restrict__ b,
                           const float* __restrict__ ms, int n) {
  long idx = (long)blockIdx.x * blockDim.x + threadIdx.x;
  if (idx >= (long)n * 128) return;
  int c = (int)(idx & 127);
  float mean = colsum[c] / (float)n;
  float var = colsq[c] / (float)n;
  float o = t[idx] - ms[c] * mean;
  float v = w[c] * o * rsqrtf(var + 1e-5f) + b[c];
  x[idx] += lrelu(v, 0.01f);
}

// ---------------------------------------------------------------------------
// Output head: dot with W_out, scalar GCN scatter, fc1, fc2
// ---------------------------------------------------------------------------
__global__ void dot1_k(const float* __restrict__ x, const float* __restrict__ w,
                       float* outv, int n) {
  long g = (long)blockIdx.x * blockDim.x + threadIdx.x;
  int node = (int)(g >> 5), l = (int)(g & 31);
  if (node >= n) return;
  const float* row = x + (long)node * 128;
  float a = 0.f;
  for (int c = l; c < 128; c += 32) a += row[c] * w[c];
  for (int off = 16; off; off >>= 1) a += __shfl_down(a, off, 32);
  if (l == 0) outv[node] = a;
}
__global__ void s_init_k(const float* lin1, const float* dinv, const float* bout,
                         float* s, int n) {
  int i = blockIdx.x * blockDim.x + threadIdx.x;
  if (i >= n) return;
  float di = dinv[i];
  s[i] = bout[0] + lin1[i] * di * di;
}
__global__ void s_edges_k(const float* lin1, const float* dinv, const int* __restrict__ ei,
                          float* s, int E) {
  int e = blockIdx.x * blockDim.x + threadIdx.x;
  if (e >= E) return;
  int a = ei[e], d = ei[E + e];
  atomicAdd(&s[d], lin1[a] * dinv[a] * dinv[d]);
}
__global__ void fc1_acc_k(const float* __restrict__ s, const float* __restrict__ W,
                          float* hfc, int n) {
  int j = threadIdx.x;  // 128 threads
  float acc = 0.f;
  for (int i = blockIdx.x; i < n; i += gridDim.x) acc += s[i] * W[(long)i * 128 + j];
  atomicAdd(&hfc[j], acc);
}
__global__ void fc1_fin_k(float* hfc, const float* b) {
  int j = threadIdx.x;
  hfc[j] = lrelu(hfc[j] + b[j], 0.01f);
}
__global__ void fc2_k(const float* hfc, const float* __restrict__ W, const float* b,
                      float* outv) {
  int k = threadIdx.x;
  float acc = b[k];
  for (int j = 0; j < 128; j++) acc += hfc[j] * W[j * 128 + k];
  outv[k] = acc;
}

// ---------------------------------------------------------------------------
// Launch sequence
// ---------------------------------------------------------------------------
extern "C" void kernel_launch(void* const* d_in, const int* in_sizes, int n_in,
                              void* d_out, int out_size, void* d_ws, size_t ws_size,
                              hipStream_t stream) {
  (void)n_in; (void)out_size; (void)ws_size;
  const int*   poi      = (const int*)  d_in[0];
  const float* dist     = (const float*)d_in[1];
  const int*   ei       = (const int*)  d_in[2];
  const float* emb      = (const float*)d_in[3];
  const float* W_in     = (const float*)d_in[4];
  const float* b_in     = (const float*)d_in[5];
  const float* gcn_W    = (const float*)d_in[6];
  const float* gcn_b    = (const float*)d_in[7];
  const float* norm_w   = (const float*)d_in[8];
  const float* norm_b   = (const float*)d_in[9];
  const float* norm_ms  = (const float*)d_in[10];
  const float* gat_W    = (const float*)d_in[11];
  const float* gat_asrc = (const float*)d_in[12];
  const float* gat_adst = (const float*)d_in[13];
  const float* gat_b    = (const float*)d_in[14];
  const float* W_out    = (const float*)d_in[15];
  const float* b_out    = (const float*)d_in[16];
  const float* fc1_W    = (const float*)d_in[17];
  const float* fc1_b    = (const float*)d_in[18];
  const float* fc2_W    = (const float*)d_in[19];
  const float* fc2_b    = (const float*)d_in[20];
  float* out = (float*)d_out;

  const int N = in_sizes[1] / 32;
  const int E = in_sizes[2] / 2;
  const int IN_K = 4128;

  char* p = (char*)d_ws;
  auto take = [&](size_t bytes) -> void* {
    void* r = (void*)p;
    p += (bytes + 255) & ~(size_t)255;
    return r;
  };
  float* dinv  = (float*)take((size_t)N * 4);
  float* xcur  = (float*)take((size_t)N * 128 * 4);
  float* hbuf  = (float*)take((size_t)N * 128 * 4);
  float* tbuf  = (float*)take((size_t)N * 128 * 4);
  float* als   = (float*)take((size_t)N * 4);
  float* ald   = (float*)take((size_t)N * 4);
  unsigned* emaxu = (unsigned*)take((size_t)N * 4);
  float* esum  = (float*)take((size_t)N * 4);
  float* lin1  = (float*)take((size_t)N * 4);
  float* svec  = (float*)take((size_t)N * 4);
  float* colsum = (float*)take(512);
  float* colsq  = (float*)take(512);
  float* hfc    = (float*)take(512);
  unsigned short* WtIn  = (unsigned short*)take((size_t)IN_K * 128 * 2);
  unsigned short* WtGcn = (unsigned short*)take((size_t)4 * 128 * 128 * 2);
  unsigned short* WtGat = (unsigned short*)take((size_t)4 * 128 * 128 * 2);

  const int T = 256;
  dim3 gN((N + T - 1) / T), gE((E + T - 1) / T);
  dim3 gNC(((long)N * 128 + T - 1) / T);
  dim3 gE32(((long)E * 32 + T - 1) / T);
  dim3 gN32(((long)N * 32 + T - 1) / T);
  dim3 gGemm((N + 127) / 128);

  // degrees -> dinv (self-loop included: init to 1)
  fill_f32_k<<<gN, T, 0, stream>>>(dinv, 1.0f, N);
  deg_edges_k<<<gE, T, 0, stream>>>(ei, dinv, E);
  rsqrt_ip_k<<<gN, T, 0, stream>>>(dinv, N);

  // weights -> bf16 transposed panels
  wtrans_k<<<(IN_K * 128 + T - 1) / T, T, 0, stream>>>(W_in, WtIn, IN_K);
  for (int l = 0; l < 4; ++l) {
    wtrans_k<<<(128 * 128 + T - 1) / T, T, 0, stream>>>(gcn_W + (long)l * 128 * 128,
                                                        WtGcn + (long)l * 128 * 128, 128);
    wtrans_k<<<(128 * 128 + T - 1) / T, T, 0, stream>>>(gat_W + (long)l * 128 * 128,
                                                        WtGat + (long)l * 128 * 128, 128);
  }

  // input layer: WMMA gemm over assembled features, then GCN scatter + leaky
  gemm_embed_k<<<gGemm, T, 0, stream>>>(poi, dist, emb, WtIn, hbuf, N);
  gcn_init_k<<<gNC, T, 0, stream>>>(hbuf, dinv, b_in, xcur, N);
  gcn_edges_k<<<gE32, T, 0, stream>>>(hbuf, xcur, ei, dinv, E);
  leaky_ip_k<<<gNC, T, 0, stream>>>(xcur, (long)N * 128, 0.01f);

  for (int l = 0; l < 4; ++l) {
    // ---- GCN sub-block ----
    gemm_x_k<<<gGemm, T, 0, stream>>>(xcur, WtGcn + (long)l * 128 * 128, hbuf, N);
    gcn_init_k<<<gNC, T, 0, stream>>>(hbuf, dinv, gcn_b + l * 128, tbuf, N);
    gcn_edges_k<<<gE32, T, 0, stream>>>(hbuf, tbuf, ei, dinv, E);
    (void)hipMemsetAsync(colsum, 0, 512, stream);
    col_acc_k<<<256, T, 0, stream>>>(tbuf, colsum, N);
    (void)hipMemsetAsync(colsq, 0, 512, stream);
    colsq_acc_k<<<256, T, 0, stream>>>(tbuf, colsum, norm_ms + l * 128, colsq, N);
    norm_res_k<<<gNC, T, 0, stream>>>(xcur, tbuf, colsum, colsq,
                                      norm_w + l * 128, norm_b + l * 128,
                                      norm_ms + l * 128, N);
    // ---- GAT sub-block ----
    gemm_x_k<<<gGemm, T, 0, stream>>>(xcur, WtGat + (long)l * 128 * 128, hbuf, N);
    gat_dots_k<<<gN32, T, 0, stream>>>(hbuf, gat_asrc + l * 128, gat_adst + l * 128,
                                       als, ald, N);
    emax_init_k<<<gN, T, 0, stream>>>(als, ald, emaxu, N);
    emax_edges_k<<<gE, T, 0, stream>>>(als, ald, emaxu, ei, E);
    esum_init_k<<<gN, T, 0, stream>>>(als, ald, emaxu, esum, N);
    esum_edges_k<<<gE, T, 0, stream>>>(als, ald, emaxu, esum, ei, E);
    gat_init_k<<<gNC, T, 0, stream>>>(hbuf, als, ald, emaxu, esum, gat_b + l * 128, tbuf, N);
    gat_edges_k<<<gE32, T, 0, stream>>>(hbuf, als, ald, emaxu, esum, ei, tbuf, E);
    (void)hipMemsetAsync(colsum, 0, 512, stream);
    col_acc_k<<<256, T, 0, stream>>>(tbuf, colsum, N);
    (void)hipMemsetAsync(colsq, 0, 512, stream);
    colsq_acc_k<<<256, T, 0, stream>>>(tbuf, colsum, norm_ms + l * 128, colsq, N);
    norm_res_k<<<gNC, T, 0, stream>>>(xcur, tbuf, colsum, colsq,
                                      norm_w + l * 128, norm_b + l * 128,
                                      norm_ms + l * 128, N);
  }

  // output GCN (C->1) + leaky
  dot1_k<<<gN32, T, 0, stream>>>(xcur, W_out, lin1, N);
  s_init_k<<<gN, T, 0, stream>>>(lin1, dinv, b_out, svec, N);
  s_edges_k<<<gE, T, 0, stream>>>(lin1, dinv, ei, svec, E);
  leaky_ip_k<<<gN, T, 0, stream>>>(svec, (long)N, 0.01f);

  // fc head
  (void)hipMemsetAsync(hfc, 0, 512, stream);
  fc1_acc_k<<<256, 128, 0, stream>>>(svec, fc1_W, hfc, N);
  fc1_fin_k<<<1, 128, 0, stream>>>(hfc, fc1_b);
  fc2_k<<<1, 128, 0, stream>>>(hfc, fc2_W, fc2_b, out);
}
